// Sparse_Res_Net_68109591380078
// MI455X (gfx1250) — matrix operand order repaired and checked by reading, not compiled, not run on hardware
//
#include <hip/hip_runtime.h>

typedef __attribute__((ext_vector_type(16))) _Float16 v16h;
typedef __attribute__((ext_vector_type(8)))  float    v8f;

#define NB 64        // batch
#define WPB 8        // waves per block in conv kernel

// ---------------------------------------------------------------------------
// Async global->LDS staging helpers (CDNA5: GLOBAL_LOAD_ASYNC_TO_LDS_*,
// tracked by ASYNCcnt). LDS aperture keeps the byte offset in the low 32 bits
// of the flat address (ISA 10.2), so truncating the
// generic pointer yields the DS address the instruction wants.
// Copies one contiguous 1KB tile: each of 32 lanes moves 16B, twice.
// ---------------------------------------------------------------------------
__device__ inline void async_copy_1kb(const _Float16* __restrict__ src,
                                      _Float16* dstLds, int lane) {
  unsigned int ldsA = (unsigned int)(unsigned long long)(dstLds + lane * 8);
  unsigned long long gA = (unsigned long long)(src + lane * 8);
  asm volatile(
      "global_load_async_to_lds_b128 %0, %1, off\n\t"
      "global_load_async_to_lds_b128 %0, %1, off offset:512"
      :: "v"(ldsA), "v"(gA) : "memory");
}
__device__ inline void wait_async() {
  asm volatile("s_wait_asynccnt 0" ::: "memory");
}

// ---------------------------------------------------------------------------
// conv0: 3->8, k=2, stride 1, pad right (offsets {0,1}), 224 -> 224, fp32
// ---------------------------------------------------------------------------
__global__ void __launch_bounds__(256) conv0_kernel(const float* __restrict__ X,
                                                    const float* __restrict__ W,
                                                    float* __restrict__ OUT) {
  int idx = blockIdx.x * 256 + threadIdx.x;
  const int total = NB * 8 * 224 * 224;
  if (idx >= total) return;
  int ow = idx % 224, t = idx / 224;
  int oh = t % 224;   t /= 224;
  int co = t % 8;     int n = t / 8;
  float acc = 0.f;
  #pragma unroll
  for (int ci = 0; ci < 3; ++ci)
    #pragma unroll
    for (int r = 0; r < 2; ++r)
      #pragma unroll
      for (int s = 0; s < 2; ++s) {
        int ih = oh + r, iw = ow + s;
        if (ih < 224 && iw < 224)
          acc += X[((n * 3 + ci) * 224 + ih) * 224 + iw] *
                 W[((co * 3 + ci) * 2 + r) * 2 + s];
      }
  OUT[idx] = acc;
}

// ---------------------------------------------------------------------------
// maxpool 4x4 stride 2 VALID: 8x224x224 -> 8x111x111
// ---------------------------------------------------------------------------
__global__ void __launch_bounds__(256) maxpool_kernel(const float* __restrict__ X,
                                                      float* __restrict__ OUT) {
  int idx = blockIdx.x * 256 + threadIdx.x;
  const int total = NB * 8 * 111 * 111;
  if (idx >= total) return;
  int ow = idx % 111, t = idx / 111;
  int oh = t % 111;   t /= 111;
  int c  = t % 8;     int n = t / 8;
  const float* base = X + ((size_t)(n * 8 + c) * 224 + oh * 2) * 224 + ow * 2;
  float m = -3.4e38f;
  #pragma unroll
  for (int i = 0; i < 4; ++i)
    #pragma unroll
    for (int j = 0; j < 4; ++j)
      m = fmaxf(m, base[i * 224 + j]);
  OUT[idx] = m;
}

// ---------------------------------------------------------------------------
// BN batch statistics: per-channel sum / sumsq via LDS reduce + global atomics
// ---------------------------------------------------------------------------
__global__ void zero_kernel(float* p, int n) {
  int i = blockIdx.x * blockDim.x + threadIdx.x;
  if (i < n) p[i] = 0.f;
}

#define BN_CHUNK 4096
__global__ void __launch_bounds__(256)
bn_stats_kernel(const float* __restrict__ X, float* __restrict__ accum,
                int C, int HW, int chunksPerPlane) {
  const int tid = threadIdx.x;
  int bid = blockIdx.x;
  int chunk = bid % chunksPerPlane;
  int plane = bid / chunksPerPlane;          // n*C + c
  int c = plane % C;
  const float* base = X + (size_t)plane * HW + (size_t)chunk * BN_CHUNK;
  int count = HW - chunk * BN_CHUNK;
  if (count > BN_CHUNK) count = BN_CHUNK;
  float s = 0.f, q = 0.f;
  for (int i = tid; i < count; i += 256) {
    float v = base[i];
    s += v; q += v * v;
  }
  __shared__ float sS[256], sQ[256];
  sS[tid] = s; sQ[tid] = q;
  __syncthreads();
  for (int off = 128; off > 0; off >>= 1) {
    if (tid < off) { sS[tid] += sS[tid + off]; sQ[tid] += sQ[tid + off]; }
    __syncthreads();
  }
  if (tid == 0) {
    atomicAdd(&accum[c],     sS[0]);
    atomicAdd(&accum[C + c], sQ[0]);
  }
}

__global__ void bn_finalize_kernel(const float* __restrict__ accum,
                                   const float* __restrict__ g,
                                   const float* __restrict__ b,
                                   float* __restrict__ scale,
                                   float* __restrict__ shift,
                                   int C, float invCnt) {
  int c = threadIdx.x;
  if (c < C) {
    float m = accum[c] * invCnt;
    float v = accum[C + c] * invCnt - m * m;
    float sc = g[c] * rsqrtf(v + 1e-4f);
    scale[c] = sc;
    shift[c] = b[c] - m * sc;
  }
}

// ---------------------------------------------------------------------------
// Pre-pack conv/linear weights (fp32, [Co, Kt]) into f16 B-tiles laid out
// exactly as the WMMA B operand expects in LDS:
//   P[(((coT*nK + kc)*2 + hh)*16 + n)*16 + s] = W[(coT*16+n)*Kt + kc*32+kl]  (0-padded),
//   kl = (s < 8) ? 8*hh + s : 16 + 8*hh + (s-8)
// Each (coT,kc) tile is a contiguous 1KB block -> async-copyable.
// ---------------------------------------------------------------------------
__global__ void __launch_bounds__(256)
pack_weights_kernel(const float* __restrict__ W, _Float16* __restrict__ P,
                    int Co, int Kt, int nK, int total) {
  int idx = blockIdx.x * 256 + threadIdx.x;
  if (idx >= total) return;
  int s  = idx & 15;
  int n  = (idx >> 4) & 15;
  int hh = (idx >> 8) & 1;
  int kl = (s < 8) ? 8 * hh + s : 16 + 8 * hh + (s - 8);
  int t  = idx >> 9;
  int kc  = t % nK;
  int coT = t / nK;
  int k  = kc * 32 + kl;
  int co = coT * 16 + n;
  P[idx] = (_Float16)((co < Co && k < Kt) ? W[(size_t)co * Kt + k] : 0.0f);
}

// ---------------------------------------------------------------------------
// Implicit-GEMM convolution via v_wmma_f32_16x16x32_f16, templated on the
// kernel window so the im2col decode uses constant (mul-shift) division.
// B tiles arrive by double-buffered async global->LDS copy issued by wave 0.
// BN-scale/shift + ReLU fused on the A operand; residual fused in epilogue.
// resMode: 0 none, 1 add raw, 2 add relu(res*resSc[c]+resSh[c])
// ---------------------------------------------------------------------------
template <int KH, int KW>
__global__ void __launch_bounds__(256)
conv_wmma_kernel(const float* __restrict__ X, const _Float16* __restrict__ Wp,
                 const float* __restrict__ scIn, const float* __restrict__ shIn,
                 const float* __restrict__ resP, const float* __restrict__ resSc,
                 const float* __restrict__ resSh, int resMode,
                 float* __restrict__ OUT,
                 int Ci, int Hi, int Wi, int Co, int Ho, int Wo,
                 int stride, int pad, int nK) {
  const int tid  = threadIdx.x;
  const int lane = tid & 31;
  const int wave = tid >> 5;
  const int mTile = blockIdx.x * WPB + wave;
  const int coT   = blockIdx.y;
  const int planeI = Hi * Wi;
  const int planeO = Ho * Wo;
  const int M  = NB * planeO;
  const int Kt = Ci * (KH * KW);

  // A-operand row handled by this lane (16-bit A 16x32 VGPR layout)
  const int mRow = lane & 15;
  const int p = mTile * 16 + mRow;
  const bool pOK = p < M;
  int n = 0, oh = 0, ow = 0;
  if (pOK) { n = p / planeO; int r = p - n * planeO; oh = r / Wo; ow = r - oh * Wo; }
  const int inBase = n * Ci * planeI;
  const int ihBase = oh * stride - pad;
  const int iwBase = ow * stride - pad;
  const int kbase = (lane < 16) ? 0 : 8;

  __shared__ alignas(32) _Float16 Bs[2][512];   // double-buffered 32x16 B tile
  const _Float16* tileBase = Wp + (size_t)coT * nK * 512;

  // prologue: stage chunk 0
  if (wave == 0) {
    async_copy_1kb(tileBase, &Bs[0][0], lane);
    wait_async();
  }
  __syncthreads();

  v8f acc = {};

  for (int kc = 0; kc < nK; ++kc) {
    const int buf = kc & 1;
    // kick off next chunk's async copy before computing on this one
    if (wave == 0 && kc + 1 < nK)
      async_copy_1kb(tileBase + (size_t)(kc + 1) * 512, &Bs[1 - buf][0], lane);

    const int kk = kc * 32;
    // gather A fragment im2col-style with fused BN+ReLU, cvt f32->f16
    v16h a;
    #pragma unroll
    for (int j = 0; j < 8; ++j) {
      int k0 = kk + kbase + ((j < 4) ? 2 * j : 16 + 2 * (j - 4));
      #pragma unroll
      for (int h = 0; h < 2; ++h) {
        int k = k0 + h;
        float v = 0.0f;
        if (pOK && k < Kt) {
          int ci = k / (KH * KW);              // constant divisor -> mul-shift
          int rs = k - ci * (KH * KW);
          int r  = rs / KW;
          int s  = rs - r * KW;
          int ih = ihBase + r;
          int iw = iwBase + s;
          if (ih >= 0 && ih < Hi && iw >= 0 && iw < Wi) {
            v = fmaxf(X[inBase + ci * planeI + ih * Wi + iw] * scIn[ci] + shIn[ci],
                      0.0f);
          }
        }
        a[2 * j + h] = (_Float16)v;
      }
    }

    // B fragment: lane holds column N=lane%16, 16 K values (table order) = 32B LDS read
    v16h b = *(const v16h*)(&Bs[buf][lane * 16]);

    acc = __builtin_amdgcn_wmma_f32_16x16x32_f16(false, a, false, b,
                                                 (short)0, acc, false, false);

    if (wave == 0 && kc + 1 < nK) wait_async();
    __syncthreads();
  }

  // epilogue: C/D layout — VGPR i holds M=i (lanes 0-15) or M=8+i (lanes 16-31)
  const int col  = lane & 15;
  const int co   = coT * 16 + col;
  const int rOff = (lane < 16) ? 0 : 8;
  if (co < Co) {
    int pe  = mTile * 16 + rOff;
    int ne  = pe / planeO;
    int rr  = pe - ne * planeO;
    int ohe = rr / Wo;
    int owe = rr - ohe * Wo;
    #pragma unroll
    for (int i = 0; i < 8; ++i) {
      if (pe < M) {
        size_t oidx = ((size_t)(ne * Co + co) * Ho + ohe) * Wo + owe;
        float val = acc[i];
        if (resMode == 1)      val += resP[oidx];
        else if (resMode == 2) val += fmaxf(resP[oidx] * resSc[co] + resSh[co], 0.0f);
        OUT[oidx] = val;
      }
      ++pe; ++owe;                                  // incremental pixel decode
      if (owe == Wo) { owe = 0; ++ohe; if (ohe == Ho) { ohe = 0; ++ne; } }
    }
  }
}

// ---------------------------------------------------------------------------
// Final linear: (64 x 6400) @ (6400 x 5) with fused BN-ReLU on features.
// Same packed-B + async-LDS pipeline; 4 waves own the 4 M tiles.
// ---------------------------------------------------------------------------
__global__ void __launch_bounds__(128)
linear_wmma_kernel(const float* __restrict__ F, const float* __restrict__ sc,
                   const float* __restrict__ sh, const _Float16* __restrict__ Wp,
                   const float* __restrict__ lb, float* __restrict__ OUT) {
  const int tid  = threadIdx.x;
  const int lane = tid & 31;
  const int mT   = tid >> 5;                 // 0..3 (batch rows mT*16..+15)
  __shared__ alignas(32) _Float16 Bs[2][512];
  v8f acc = {};
  const int row  = lane & 15;
  const int nrow = mT * 16 + row;
  const int kbase = (lane < 16) ? 0 : 8;
  const int nK = 200;                        // 6400 / 32

  if (mT == 0) {
    async_copy_1kb(Wp, &Bs[0][0], lane);
    wait_async();
  }
  __syncthreads();

  for (int kc = 0; kc < nK; ++kc) {
    const int buf = kc & 1;
    if (mT == 0 && kc + 1 < nK)
      async_copy_1kb(Wp + (size_t)(kc + 1) * 512, &Bs[1 - buf][0], lane);

    const int kk = kc * 32;
    v16h a;
    #pragma unroll
    for (int j = 0; j < 8; ++j) {
      int k0 = kk + kbase + ((j < 4) ? 2 * j : 16 + 2 * (j - 4));
      #pragma unroll
      for (int h = 0; h < 2; ++h) {
        int k = k0 + h;
        int c = k / 100;                     // feature k = c*100 + h*10 + w
        float v = fmaxf(F[(size_t)nrow * 6400 + k] * sc[c] + sh[c], 0.0f);
        a[2 * j + h] = (_Float16)v;
      }
    }
    v16h b = *(const v16h*)(&Bs[buf][lane * 16]);
    acc = __builtin_amdgcn_wmma_f32_16x16x32_f16(false, a, false, b,
                                                 (short)0, acc, false, false);
    if (mT == 0 && kc + 1 < nK) wait_async();
    __syncthreads();
  }
  const int col  = lane & 15;
  const int rOff = (lane < 16) ? 0 : 8;
  if (col < 5) {
    #pragma unroll
    for (int i = 0; i < 8; ++i) {
      int nn = mT * 16 + rOff + i;
      OUT[nn * 5 + col] = acc[i] + lb[col];
    }
  }
}

// ---------------------------------------------------------------------------
// Host orchestration
// ---------------------------------------------------------------------------
struct BlkCfg { int ci, co, st, first, resconv; };
static const BlkCfg BCFG[9] = {
  {8, 8, 1, 1, 0}, {8, 8, 1, 0, 0}, {8, 8, 1, 0, 0},
  {8, 16, 2, 1, 1}, {16, 16, 1, 0, 0},
  {16, 24, 2, 1, 1}, {24, 24, 1, 0, 0},
  {24, 32, 2, 1, 1}, {32, 32, 1, 0, 0}};

static void run_bn(const float* T, int C, int HW, const float* g, const float* b,
                   float* accum, float* scale, float* shift, hipStream_t stream) {
  zero_kernel<<<(2 * C + 127) / 128, 128, 0, stream>>>(accum, 2 * C);
  int chunks = (HW + BN_CHUNK - 1) / BN_CHUNK;
  bn_stats_kernel<<<NB * C * chunks, 256, 0, stream>>>(T, accum, C, HW, chunks);
  bn_finalize_kernel<<<1, 64, 0, stream>>>(accum, g, b, scale, shift, C,
                                           1.0f / ((float)NB * (float)HW));
}

static void run_conv(const float* X, const float* Wt, _Float16* Wp,
                     const float* sIn, const float* hIn,
                     const float* resP, const float* rs, const float* rh, int resMode,
                     float* OUT, int Ci, int Hi, int Wi, int Co, int Ho, int Wo,
                     int kh, int stride, int pad, hipStream_t stream) {
  int Kt = Ci * kh * kh;
  int nK = (Kt + 31) / 32;
  int coTiles = (Co + 15) / 16;
  int total = coTiles * nK * 512;
  pack_weights_kernel<<<(total + 255) / 256, 256, 0, stream>>>(Wt, Wp, Co, Kt,
                                                               nK, total);
  int M = NB * Ho * Wo;
  int mTiles = (M + 15) / 16;
  dim3 grid((mTiles + WPB - 1) / WPB, coTiles);
  if (kh == 3)
    conv_wmma_kernel<3, 3><<<grid, WPB * 32, 0, stream>>>(
        X, Wp, sIn, hIn, resP, rs, rh, resMode, OUT,
        Ci, Hi, Wi, Co, Ho, Wo, stride, pad, nK);
  else
    conv_wmma_kernel<4, 4><<<grid, WPB * 32, 0, stream>>>(
        X, Wp, sIn, hIn, resP, rs, rh, resMode, OUT,
        Ci, Hi, Wi, Co, Ho, Wo, stride, pad, nK);
}

extern "C" void kernel_launch(void* const* d_in, const int* in_sizes, int n_in,
                              void* d_out, int out_size, void* d_ws, size_t ws_size,
                              hipStream_t stream) {
  (void)in_sizes; (void)n_in; (void)out_size; (void)ws_size;

  int idx = 0;
  const float* x        = (const float*)d_in[idx++];
  const float* w_conv0  = (const float*)d_in[idx++];
  const float *g1[9], *b1[9], *w1[9], *g2[9], *b2[9], *w2[9], *wr[9];
  for (int i = 0; i < 9; ++i) {
    g1[i] = (const float*)d_in[idx++]; b1[i] = (const float*)d_in[idx++];
    w1[i] = (const float*)d_in[idx++]; g2[i] = (const float*)d_in[idx++];
    b2[i] = (const float*)d_in[idx++]; w2[i] = (const float*)d_in[idx++];
    wr[i] = BCFG[i].resconv ? (const float*)d_in[idx++] : nullptr;
  }
  const float* bn_f_g = (const float*)d_in[idx++];
  const float* bn_f_b = (const float*)d_in[idx++];
  const float* w_last = (const float*)d_in[idx++];
  const float* bn_l_g = (const float*)d_in[idx++];
  const float* bn_l_b = (const float*)d_in[idx++];
  const float* lin_w  = (const float*)d_in[idx++];
  const float* lin_b  = (const float*)d_in[idx++];

  // workspace layout (floats; every offset a multiple of 64 -> 256B aligned)
  float* ws = (float*)d_ws;
  size_t o = 0;
  float* A   = ws + o; o += (size_t)NB * 8 * 224 * 224;   // conv0 out
  float* Xb0 = ws + o; o += (size_t)NB * 8 * 111 * 111;   // ping
  float* Xb1 = ws + o; o += (size_t)NB * 8 * 111 * 111;   // pong
  float* T   = ws + o; o += (size_t)NB * 8 * 111 * 111;   // conv1 temp
  float* R   = ws + o; o += (size_t)NB * 16 * 55 * 55;    // residual-conv temp
  float* CL  = ws + o; o += (size_t)NB * 64 * 10 * 10;    // conv_last out
  float* ACC = ws + o; o += 128;                          // bn sum/sumsq
  float* S1  = ws + o; o += 64;
  float* H1  = ws + o; o += 64;
  float* S2  = ws + o; o += 64;
  float* H2  = ws + o; o += 64;
  _Float16* Wp = (_Float16*)(ws + o); o += 51200;         // packed B (102400 h)

  // stem
  {
    int total = NB * 8 * 224 * 224;
    conv0_kernel<<<(total + 255) / 256, 256, 0, stream>>>(x, w_conv0, A);
    total = NB * 8 * 111 * 111;
    maxpool_kernel<<<(total + 255) / 256, 256, 0, stream>>>(A, Xb0);
  }

  float* xb[2] = {Xb0, Xb1};
  int cur = 0;
  int hw = 111;

  for (int i = 0; i < 9; ++i) {
    const BlkCfg c = BCFG[i];
    int Ho = (c.st == 1) ? hw : (hw - 3) / 2 + 1;
    int pad1 = (c.st == 1) ? 1 : 0;

    // bn1 over current x, then conv1 (BN+ReLU fused on operand load)
    run_bn(xb[cur], c.ci, hw * hw, g1[i], b1[i], ACC, S1, H1, stream);
    run_conv(xb[cur], w1[i], Wp, S1, H1, nullptr, nullptr, nullptr, 0,
             T, c.ci, hw, hw, c.co, Ho, Ho, 3, c.st, pad1, stream);

    // projection residual: conv(bn_relu(x)) stride-2 VALID
    if (c.resconv)
      run_conv(xb[cur], wr[i], Wp, S1, H1, nullptr, nullptr, nullptr, 0,
               R, c.ci, hw, hw, c.co, Ho, Ho, 3, c.st, 0, stream);

    // bn2 over conv1 output, then conv2 with fused residual add
    run_bn(T, c.co, Ho * Ho, g2[i], b2[i], ACC, S2, H2, stream);

    const float* resP; const float *rs = nullptr, *rh = nullptr; int rm;
    if (c.resconv)    { resP = R;       rm = 1; }                  // projected
    else if (c.first) { resP = xb[cur]; rm = 2; rs = S1; rh = H1; }// y=bnrelu(x)
    else              { resP = xb[cur]; rm = 1; }                  // identity x
    run_conv(T, w2[i], Wp, S2, H2, resP, rs, rh, rm,
             xb[1 - cur], c.co, Ho, Ho, c.co, Ho, Ho, 3, 1, 1, stream);

    cur ^= 1;
    hw = Ho;
  }

  // final BN + conv_last (32->64, k=4 VALID: 13 -> 10)
  run_bn(xb[cur], 32, hw * hw, bn_f_g, bn_f_b, ACC, S1, H1, stream);
  run_conv(xb[cur], w_last, Wp, S1, H1, nullptr, nullptr, nullptr, 0,
           CL, 32, 13, 13, 64, 10, 10, 4, 1, 0, stream);

  // bn_l stats, then fused BN-ReLU + linear via WMMA
  run_bn(CL, 64, 100, bn_l_g, bn_l_b, ACC, S2, H2, stream);
  // pack lin_w (5 x 6400) into B tiles: coTiles=1, nK=200
  {
    int total = 1 * 200 * 512;
    pack_weights_kernel<<<(total + 255) / 256, 256, 0, stream>>>(lin_w, Wp, 5,
                                                                 6400, 200, total);
  }
  linear_wmma_kernel<<<1, 128, 0, stream>>>(CL, S2, H2, Wp, lin_b,
                                            (float*)d_out);
}
